// AttentionalCopula_62732292326012
// MI455X (gfx1250) — matrix operand order repaired and checked by reading, not compiled, not run on hardware
//
#include <hip/hip_runtime.h>

typedef __attribute__((ext_vector_type(16))) _Float16 v16h;
typedef __attribute__((ext_vector_type(8)))  _Float16 v8h;
typedef __attribute__((ext_vector_type(8)))  float    v8f;

// ---------------------------------------------------------------------------
// Generic batched WMMA GEMM:  C = act(A @ Bsw + bias)
//   A   : f16 row-major [rows x K], per-batch stride sA (halves)
//   Bsw : f16, fragment-major: frag(kt,nt) = 32 lanes x 16 halves contiguous
//         (lane -> column n = nt*16 + lane%16 ; halves j -> k = kt*32 + (lane<16?0:8) + (j<8?j:j+8))
//   C   : row-major, f16 or f32
// Block = 256 threads (8 waves). Wave w owns rows [bx*128+w*16, +16).
// Each wave computes NT 16x16 tiles along N (ntile0 = by*NT).
// Software-pipelined: fragments for k-step k+1 are loaded before the WMMAs of
// k-step k issue, so load clauses overlap matrix ops (partial loadcnt waits).
// ---------------------------------------------------------------------------
template<int NT>
__global__ __launch_bounds__(256) void gemm_kernel(
    const _Float16* __restrict__ A, int ldA, long sA,
    const _Float16* __restrict__ Bsw, long sB,
    const float* __restrict__ bias, int sBias,
    void* __restrict__ C, int ldC, long sC,
    int K, int N, int relu, int outf32)
{
  const int lane = threadIdx.x & 31;
  const int wave = threadIdx.x >> 5;
  const int z = blockIdx.z;
  A   += (long)z * sA;
  Bsw += (long)z * sB;
  const float* bp = bias + (long)z * sBias;

  const int row0   = blockIdx.x * 128 + wave * 16;
  const int ntile0 = blockIdx.y * NT;
  const int mloc   = lane & 15;
  const int koff   = (lane < 16) ? 0 : 8;
  const _Float16* Arow = A + (long)(row0 + mloc) * ldA;
  const int ntN = N >> 4;
  const int K32 = K >> 5;

  v8f zero = {0.f,0.f,0.f,0.f,0.f,0.f,0.f,0.f};
  v8f acc[NT];
#pragma unroll
  for (int nt = 0; nt < NT; ++nt) acc[nt] = zero;

  // fragment load: A = two contiguous 16B chunks (16-bit A layout), B = 32B/lane
  auto LD = [&](int kt, v16h& a, v16h* b) {
    const _Float16* ap = Arow + kt * 32 + koff;
    v8h alo = *(const v8h*)(ap);
    v8h ahi = *(const v8h*)(ap + 16);
#pragma unroll
    for (int j = 0; j < 8; ++j) { a[j] = alo[j]; a[j + 8] = ahi[j]; }
    const _Float16* bbase = Bsw + ((long)kt * ntN + ntile0) * 512 + lane * 16;
#pragma unroll
    for (int nt = 0; nt < NT; ++nt) b[nt] = *(const v16h*)(bbase + nt * 512);
  };
  auto MM = [&](v16h& a, v16h* b) {
#pragma unroll
    for (int nt = 0; nt < NT; ++nt)
      acc[nt] = __builtin_amdgcn_wmma_f32_16x16x32_f16(
          false, a, false, b[nt], (short)0, acc[nt], false, false);
  };

  v16h aP, aN;
  v16h bP[NT], bN[NT];
  LD(0, aP, bP);
  int kt = 0;
  for (; kt + 2 <= K32; kt += 2) {
    LD(kt + 1, aN, bN);             // prefetch next k-step into second buffer
    MM(aP, bP);
    if (kt + 2 < K32) LD(kt + 2, aP, bP);
    MM(aN, bN);
  }
  if (kt < K32) MM(aP, bP);         // odd K32 tail (already loaded)

  // C/D layout: VGPR r, lane l -> (M = r + (l<16?0:8), N = l%16)
  const int rbase = row0 + ((lane < 16) ? 0 : 8);
#pragma unroll
  for (int nt = 0; nt < NT; ++nt) {
    const int n = (ntile0 + nt) * 16 + mloc;
    const float bv = bp[n];
#pragma unroll
    for (int r = 0; r < 8; ++r) {
      float v = acc[nt][r] + bv;
      if (relu) v = fmaxf(v, 0.f);
      long idx = (long)z * sC + (long)(rbase + r) * ldC + n;
      if (outf32) ((float*)C)[idx] = v;
      else        ((_Float16*)C)[idx] = (_Float16)v;
    }
  }
}

// ---------------------------------------------------------------------------
// Swizzle f32 row-major W[Ksrc x N] -> f16 fragment-major (Kp padded, mult 32)
// ---------------------------------------------------------------------------
__global__ void swizzle_kernel(const float* __restrict__ src, long sBatch,
                               _Float16* __restrict__ dst,
                               int Ksrc, int Kp, int N)
{
  const int lane = threadIdx.x & 31;
  const int frag = blockIdx.x * 8 + (threadIdx.x >> 5);
  const int ntN = N >> 4;
  const int nf = (Kp >> 5) * ntN;
  if (frag >= nf) return;
  const int z = blockIdx.y;
  src += (long)z * sBatch;
  dst += (long)z * nf * 512;
  const int kt = frag / ntN, nt = frag - kt * ntN;
  const int n  = nt * 16 + (lane & 15);
  const int k0 = kt * 32 + ((lane & 16) ? 8 : 0);
  v16h o;
#pragma unroll
  for (int j = 0; j < 16; ++j) {
    int k = k0 + (j < 8 ? j : j + 8);
    float v = (k < Ksrc) ? src[(long)k * N + n] : 0.f;
    o[j] = (_Float16)v;
  }
  *((v16h*)(dst + (long)frag * 512) + lane) = o;
}

// ---------------------------------------------------------------------------
// Gather MLP input rows.  row r = ((b*8+s)*2+n)*128+m ; cols 0..255 = encoded,
// col 256 = true_u, cols 257..287 = 0 (zero-padded K).
// ---------------------------------------------------------------------------
__global__ void gather_x_kernel(const float* __restrict__ enc,
                                const float* __restrict__ tu,
                                const int* __restrict__ lidx,
                                const int* __restrict__ ridx,
                                const int* __restrict__ ip,
                                _Float16* __restrict__ Xg)
{
  const int r = blockIdx.x, tid = threadIdx.x;
  const int m = r & 127, n = (r >> 7) & 1, s = (r >> 8) & 7, b = r >> 11;
  const int ii = ip[0];
  const int t = (n == 0) ? lidx[m] : ((s < ii) ? m : ridx[m]);
  const long base = (((long)(b * 8 + s)) * 256 + t) * 256;
  Xg[(long)r * 288 + tid] = (_Float16)enc[base + tid];
  if (tid < 32) {
    _Float16 v = (_Float16)0.f;
    if (tid == 0) v = (_Float16)tu[(long)(b * 8 + s) * 256 + t];
    Xg[(long)r * 288 + 256 + tid] = v;
  }
}

__global__ void gather_pred_kernel(const float* __restrict__ enc,
                                   const int* __restrict__ midx,
                                   const int* __restrict__ ip,
                                   _Float16* __restrict__ P)
{
  const int r = blockIdx.x, tid = threadIdx.x;
  const int b = r >> 7, m = r & 127;
  const int ii = ip[0];
  const int t = midx[m];
  P[(long)r * 256 + tid] =
      (_Float16)enc[(((long)(b * 8 + ii)) * 256 + t) * 256 + tid];
}

// ---------------------------------------------------------------------------
// Attention (reference semantics: softmax over n (size 2) per (h,s), then sum
// over all (s,n)).  KV layout: [batch=(kind*16+l*8+h)][row=16384][64] f16.
// One block per (b,m); y = att_value + att (residual included).
// ---------------------------------------------------------------------------
__global__ __launch_bounds__(128) void attention_kernel(
    const float* __restrict__ av, const _Float16* __restrict__ KV,
    int l, float* __restrict__ y)
{
  __shared__ float q[512];
  __shared__ float sc[128];
  __shared__ float wgt[128];
  __shared__ _Float16 snk[8192];
  __shared__ _Float16 snv[8192];
  const int bm = blockIdx.x, tid = threadIdx.x;
  const int b = bm >> 7, m = bm & 127;
#pragma unroll
  for (int k = 0; k < 4; ++k) q[tid + k * 128] = av[(long)bm * 512 + tid + k * 128];
  for (int idx = tid; idx < 8192; idx += 128) {
    const int j = idx >> 9, c = idx & 511;
    const int h = c >> 6, d = c & 63;
    const int s = j >> 1, n = j & 1;
    const long r = ((long)((b * 8 + s) * 2 + n)) * 128 + m;
    snk[idx] = KV[((long)(l * 8 + h) * 16384 + r) * 64 + d];
    snv[idx] = KV[((long)(16 + l * 8 + h) * 16384 + r) * 64 + d];
  }
  __syncthreads();
  {
    const int h = tid >> 4, j = tid & 15;
    float s = 0.f;
    for (int d = 0; d < 64; ++d) s += q[h * 64 + d] * (float)snk[j * 512 + h * 64 + d];
    sc[tid] = s * 0.125f;  // DK^-0.5
  }
  __syncthreads();
  {
    const int base = tid & ~1;
    const float a0 = sc[base], a1 = sc[base + 1];
    const float mx = fmaxf(a0, a1);
    const float e0 = expf(a0 - mx), e1 = expf(a1 - mx);
    wgt[tid] = ((tid & 1) ? e1 : e0) / (e0 + e1);
  }
  __syncthreads();
#pragma unroll
  for (int k = 0; k < 4; ++k) {
    const int c = tid + k * 128;
    const int h = c >> 6;
    float s = 0.f;
#pragma unroll
    for (int j = 0; j < 16; ++j) s += wgt[h * 16 + j] * (float)snv[j * 512 + c];
    y[(long)bm * 512 + c] = av[(long)bm * 512 + c] + s;
  }
}

// ---------------------------------------------------------------------------
// LayerNorm over 512, optional second residual input; writes f32 + f16 copies
// ---------------------------------------------------------------------------
__global__ __launch_bounds__(256) void ln_kernel(
    const float* __restrict__ in1, const float* __restrict__ in2,
    const float* __restrict__ g, const float* __restrict__ bb,
    float* __restrict__ o32, _Float16* __restrict__ o16)
{
  __shared__ float s1[256], s2[256];
  const int row = blockIdx.x, tid = threadIdx.x;
  const long base = (long)row * 512;
  float x0 = in1[base + tid];
  float x1 = in1[base + tid + 256];
  if (in2) { x0 += in2[base + tid]; x1 += in2[base + tid + 256]; }
  s1[tid] = x0 + x1;
  s2[tid] = x0 * x0 + x1 * x1;
  __syncthreads();
  for (int off = 128; off > 0; off >>= 1) {
    if (tid < off) { s1[tid] += s1[tid + off]; s2[tid] += s2[tid + off]; }
    __syncthreads();
  }
  const float mu  = s1[0] * (1.f / 512.f);
  const float var = s2[0] * (1.f / 512.f) - mu * mu;
  const float rs  = rsqrtf(var + 1e-5f);
  const float y0 = (x0 - mu) * rs * g[tid] + bb[tid];
  const float y1 = (x1 - mu) * rs * g[tid + 256] + bb[tid + 256];
  o32[base + tid] = y0;       o32[base + tid + 256] = y1;
  o16[base + tid] = (_Float16)y0;
  o16[base + tid + 256] = (_Float16)y1;
}

// ---------------------------------------------------------------------------
extern "C" void kernel_launch(void* const* d_in, const int* in_sizes, int n_in,
                              void* d_out, int out_size, void* d_ws, size_t ws_size,
                              hipStream_t stream) {
  (void)in_sizes; (void)n_in; (void)out_size; (void)ws_size;
  const float* encoded  = (const float*)d_in[0];
  const float* true_u   = (const float*)d_in[1];
  const int*   mid_idx  = (const int*)d_in[2];
  const int*   left_idx = (const int*)d_in[3];
  const int*   right_idx= (const int*)d_in[4];
  const int*   ip       = (const int*)d_in[5];
  const float* ds_W = (const float*)d_in[6];
  const float* ds_b = (const float*)d_in[7];
  const float* kW1 = (const float*)d_in[8];  const float* kb1 = (const float*)d_in[9];
  const float* kW2 = (const float*)d_in[10]; const float* kb2 = (const float*)d_in[11];
  const float* kW3 = (const float*)d_in[12]; const float* kb3 = (const float*)d_in[13];
  const float* vW1 = (const float*)d_in[14]; const float* vb1 = (const float*)d_in[15];
  const float* vW2 = (const float*)d_in[16]; const float* vb2 = (const float*)d_in[17];
  const float* vW3 = (const float*)d_in[18]; const float* vb3 = (const float*)d_in[19];
  const float* ln1_g = (const float*)d_in[20]; const float* ln1_b = (const float*)d_in[21];
  const float* ln2_g = (const float*)d_in[22]; const float* ln2_b = (const float*)d_in[23];
  const float* ffW1 = (const float*)d_in[24]; const float* ffb1 = (const float*)d_in[25];
  const float* ffW2 = (const float*)d_in[26]; const float* ffb2 = (const float*)d_in[27];
  const float* deW1 = (const float*)d_in[28]; const float* deb1 = (const float*)d_in[29];
  const float* deW2 = (const float*)d_in[30]; const float* deb2 = (const float*)d_in[31];
  const float* deW3 = (const float*)d_in[32]; const float* deb3 = (const float*)d_in[33];

  // ---- workspace carve (aligned 256B) ----
  size_t off = 0;
  auto take = [&](size_t bytes) -> char* {
    char* p = (char*)d_ws + off;
    off += (bytes + 255) & ~(size_t)255;
    return p;
  };
  const long NR = 16384;                       // gathered rows
  _Float16* Xg     = (_Float16*)take(NR * 288 * 2);
  _Float16* predg  = (_Float16*)take(1024L * 256 * 2);
  _Float16* W1sw   = (_Float16*)take(32L * 144 * 512 * 2);
  _Float16* W2sw   = (_Float16*)take(32L * 128 * 512 * 2);
  _Float16* W3sw   = (_Float16*)take(32L * 32 * 512 * 2);
  _Float16* ffW1sw = (_Float16*)take(2L * 512 * 512 * 2);
  _Float16* ffW2sw = (_Float16*)take(2L * 512 * 512 * 2);
  _Float16* deW1sw = (_Float16*)take(256L * 512 * 2);
  _Float16* deW2sw = (_Float16*)take(128L * 512 * 2);
  _Float16* deW3sw = (_Float16*)take(256L * 512 * 2);
  _Float16* dsWsw  = (_Float16*)take(256L * 512 * 2);
  _Float16* H1     = (_Float16*)take(2L * NR * 256 * 2);   // chunk of 2 batches
  _Float16* H2     = (_Float16*)take(2L * NR * 256 * 2);
  _Float16* KV     = (_Float16*)take(32L * NR * 64 * 2);
  float*    avf32  = (float*)take(1024L * 512 * 4);
  _Float16* avf16  = (_Float16*)take(1024L * 512 * 2);
  float*    ybuf   = (float*)take(1024L * 512 * 4);
  float*    xf32   = (float*)take(1024L * 512 * 4);
  _Float16* xf16   = (_Float16*)take(1024L * 512 * 2);
  _Float16* ffh1   = (_Float16*)take(1024L * 512 * 2);
  float*    ffo    = (float*)take(1024L * 512 * 4);
  _Float16* dd1    = (_Float16*)take(1024L * 256 * 2);
  _Float16* dd2    = (_Float16*)take(1024L * 256 * 2);

  // ---- weight swizzles ----
  swizzle_kernel<<<dim3(18,16),256,0,stream>>>(kW1, 257L*256, W1sw,              257, 288, 256);
  swizzle_kernel<<<dim3(18,16),256,0,stream>>>(vW1, 257L*256, W1sw + 16L*144*512,257, 288, 256);
  swizzle_kernel<<<dim3(16,16),256,0,stream>>>(kW2, 256L*256, W2sw,              256, 256, 256);
  swizzle_kernel<<<dim3(16,16),256,0,stream>>>(vW2, 256L*256, W2sw + 16L*128*512,256, 256, 256);
  swizzle_kernel<<<dim3(4,16), 256,0,stream>>>(kW3, 256L*64,  W3sw,              256, 256, 64);
  swizzle_kernel<<<dim3(4,16), 256,0,stream>>>(vW3, 256L*64,  W3sw + 16L*32*512, 256, 256, 64);
  swizzle_kernel<<<dim3(64,2), 256,0,stream>>>(ffW1, 512L*512, ffW1sw, 512, 512, 512);
  swizzle_kernel<<<dim3(64,2), 256,0,stream>>>(ffW2, 512L*512, ffW2sw, 512, 512, 512);
  swizzle_kernel<<<dim3(32,1), 256,0,stream>>>(deW1, 0, deW1sw, 512, 512, 256);
  swizzle_kernel<<<dim3(16,1), 256,0,stream>>>(deW2, 0, deW2sw, 256, 256, 256);
  swizzle_kernel<<<dim3(32,1), 256,0,stream>>>(deW3, 0, deW3sw, 256, 256, 512);
  swizzle_kernel<<<dim3(32,1), 256,0,stream>>>(ds_W, 0, dsWsw,  256, 256, 512);

  // ---- input gathers ----
  gather_x_kernel<<<16384,256,0,stream>>>(encoded, true_u, left_idx, right_idx, ip, Xg);
  gather_pred_kernel<<<1024,256,0,stream>>>(encoded, mid_idx, ip, predg);

  // ---- Phase A: 32 (kind,l,h) MLP batches, chunks of 2 ----
  for (int c = 0; c < 16; ++c) {
    const int g0 = c * 2;
    const int kind = g0 >> 4;
    const int lh0 = g0 & 15;
    const float* b1 = (kind ? vb1 : kb1) + lh0 * 256;
    const float* b2 = (kind ? vb2 : kb2) + lh0 * 256;
    const float* b3 = (kind ? vb3 : kb3) + lh0 * 64;
    gemm_kernel<8><<<dim3(128,2,2),256,0,stream>>>(
        Xg, 288, 0, W1sw + (long)g0*144*512, 144L*512, b1, 256,
        H1, 256, NR*256L, 288, 256, 1, 0);
    gemm_kernel<8><<<dim3(128,2,2),256,0,stream>>>(
        H1, 256, NR*256L, W2sw + (long)g0*128*512, 128L*512, b2, 256,
        H2, 256, NR*256L, 256, 256, 1, 0);
    gemm_kernel<4><<<dim3(128,1,2),256,0,stream>>>(
        H2, 256, NR*256L, W3sw + (long)g0*32*512, 32L*512, b3, 64,
        KV + (long)g0*NR*64, 64, NR*64L, 256, 64, 0, 0);
  }

  // ---- Phase B ----
  // att_value = pred @ ds_W + ds_b  (f32 out)
  gemm_kernel<8><<<dim3(8,4,1),256,0,stream>>>(
      predg, 256, 0, dsWsw, 0, ds_b, 0, avf32, 512, 0, 256, 512, 0, 1);

  for (int l = 0; l < 2; ++l) {
    attention_kernel<<<1024,128,0,stream>>>(avf32, KV, l, ybuf);
    ln_kernel<<<1024,256,0,stream>>>(ybuf, nullptr, ln1_g + l*512, ln1_b + l*512,
                                     xf32, xf16);
    gemm_kernel<8><<<dim3(8,4,1),256,0,stream>>>(
        xf16, 512, 0, ffW1sw + (long)l*512*512, 0, ffb1 + l*512, 0,
        ffh1, 512, 0, 512, 512, 1, 0);
    gemm_kernel<8><<<dim3(8,4,1),256,0,stream>>>(
        ffh1, 512, 0, ffW2sw + (long)l*512*512, 0, ffb2 + l*512, 0,
        ffo, 512, 0, 512, 512, 0, 1);
    ln_kernel<<<1024,256,0,stream>>>(xf32, ffo, ln2_g + l*512, ln2_b + l*512,
                                     avf32, avf16);
  }

  // ---- decoder ----
  gemm_kernel<8><<<dim3(8,2,1),256,0,stream>>>(
      avf16, 512, 0, deW1sw, 0, deb1, 0, dd1, 256, 0, 512, 256, 1, 0);
  gemm_kernel<8><<<dim3(8,2,1),256,0,stream>>>(
      dd1, 256, 0, deW2sw, 0, deb2, 0, dd2, 256, 0, 256, 256, 1, 0);
  gemm_kernel<8><<<dim3(8,4,1),256,0,stream>>>(
      dd2, 256, 0, deW3sw, 0, deb3, 0, (float*)d_out, 512, 0, 256, 512, 0, 1);
}
